// AttentionBase_34789235098377
// MI455X (gfx1250) — matrix operand order repaired and checked by reading, not compiled, not run on hardware
//
#include <hip/hip_runtime.h>
#include <cstdint>
#include <cstddef>

#define WORD_DIM    100
#define TITLE_LEN   8192
#define NUM_HEADERS 4096
#define KPAD        128

typedef __attribute__((ext_vector_type(16))) __bf16 v16bf;
typedef __attribute__((ext_vector_type(8)))  float  v8f;

union FragBF { v16bf v; unsigned u[8]; };

__device__ __forceinline__ unsigned short f2bf_rne(float f) {
  unsigned u = __float_as_uint(f);
  u += 0x7FFFu + ((u >> 16) & 1u);
  return (unsigned short)(u >> 16);
}
// monotone (order-preserving) uint encoding of float, for atomicMax on floats
__device__ __forceinline__ unsigned enc_ord(float f) {
  unsigned u = __float_as_uint(f);
  return (u & 0x80000000u) ? ~u : (u | 0x80000000u);
}
__device__ __forceinline__ float dec_ord(unsigned u) {
  return (u & 0x80000000u) ? __uint_as_float(u ^ 0x80000000u)
                           : __uint_as_float(~u);
}

// ---- prep: headers*w_cq -> bf16 padded [4096][128]; rowterm = h.w_c + b_c ----
__global__ void prep_headers_kernel(const float* __restrict__ data,
                                    const float* __restrict__ w_cq,
                                    const float* __restrict__ w_c,
                                    const float* __restrict__ b_c,
                                    unsigned short* __restrict__ hwB,
                                    float* __restrict__ rowterm,
                                    unsigned* __restrict__ rowMaxU) {
  const int lane = threadIdx.x & 31;
  const int wid  = threadIdx.x >> 5;
  const int m = blockIdx.x * 8 + wid;
  float dot = 0.f;
  for (int k = lane; k < KPAD; k += 32) {
    float hw = 0.f;
    if (k < WORD_DIM) {
      const float h = data[m * 500 + k];
      hw = h * w_cq[k];
      dot += h * w_c[k];
    }
    hwB[m * KPAD + k] = f2bf_rne(hw);
  }
  #pragma unroll
  for (int o = 1; o < 32; o <<= 1) dot += __shfl_xor(dot, o, 32);
  if (lane == 0) { rowterm[m] = dot + b_c[0]; rowMaxU[m] = 0u; }
}

// ---- prep: title -> bf16 padded [8192][128]; colterm = t.w_q + b_q ----
__global__ void prep_title_kernel(const float* __restrict__ title,
                                  const float* __restrict__ w_q,
                                  const float* __restrict__ b_q,
                                  unsigned short* __restrict__ tbB,
                                  float* __restrict__ colterm,
                                  unsigned* __restrict__ colMaxU) {
  const int lane = threadIdx.x & 31;
  const int wid  = threadIdx.x >> 5;
  const int c = blockIdx.x * 8 + wid;
  float dot = 0.f;
  for (int k = lane; k < KPAD; k += 32) {
    float tv = 0.f;
    if (k < WORD_DIM) { tv = title[c * WORD_DIM + k]; dot += tv * w_q[k]; }
    tbB[c * KPAD + k] = f2bf_rne(tv);
  }
  #pragma unroll
  for (int o = 1; o < 32; o <<= 1) dot += __shfl_xor(dot, o, 32);
  if (lane == 0) { colterm[c] = dot + b_q[0]; colMaxU[c] = 0u; }
}

// ---- fused sim GEMM (bf16 WMMA) + row/col max reduction, sim never stored ----
__global__ void __launch_bounds__(256) sim_gemm_max_kernel(
    const unsigned short* __restrict__ hwB,
    const unsigned short* __restrict__ tbB,
    const float* __restrict__ rowterm,
    const float* __restrict__ colterm,
    const float* __restrict__ b_cq,
    unsigned* __restrict__ rowMaxU,
    unsigned* __restrict__ colMaxU) {
  const int lane = threadIdx.x & 31;
  const int wid  = threadIdx.x >> 5;
  const int wm = wid & 1;            // 2 waves along M
  const int wn = wid >> 1;           // 4 waves along N
  const int mwave = blockIdx.y * 128 + wm * 64;
  const int nwave = blockIdx.x * 128 + wn * 32;
  const int hf  = lane >> 4;
  const int l16 = lane & 15;

  const v8f vzero = {0.f, 0.f, 0.f, 0.f, 0.f, 0.f, 0.f, 0.f};
  v8f acc[4][2];
  #pragma unroll
  for (int i = 0; i < 4; ++i)
    #pragma unroll
    for (int j = 0; j < 2; ++j) acc[i][j] = vzero;

  #pragma unroll
  for (int kk = 0; kk < KPAD; kk += 32) {
    FragBF bfrag[2];
    #pragma unroll
    for (int j = 0; j < 2; ++j) {
      // B 32x16 bf16: lane = N, K = (v<4?0:16) + 8*half + 2*(v&3)
      const unsigned short* bp =
          tbB + (size_t)(nwave + j * 16 + l16) * KPAD + kk + 8 * hf;
      #pragma unroll
      for (int v = 0; v < 8; ++v)
        bfrag[j].u[v] = *(const unsigned*)(bp + ((v < 4) ? 0 : 16) + 2 * (v & 3));
    }
    #pragma unroll
    for (int i = 0; i < 4; ++i) {
      // A 16x32 bf16: lane = M, K = (v<4?0:16) + 8*half + 2*(v&3)
      FragBF afrag;
      const unsigned short* ap =
          hwB + (size_t)(mwave + i * 16 + l16) * KPAD + kk + 8 * hf;
      #pragma unroll
      for (int v = 0; v < 8; ++v)
        afrag.u[v] = *(const unsigned*)(ap + ((v < 4) ? 0 : 16) + 2 * (v & 3));
      #pragma unroll
      for (int j = 0; j < 2; ++j)
        acc[i][j] = __builtin_amdgcn_wmma_f32_16x16x32_bf16(
            false, afrag.v, false, bfrag[j].v, (short)0, acc[i][j], false, false);
    }
  }

  // epilogue: sim = acc + rowterm + colterm + b_cq; fused row/col max
  const float bcq = b_cq[0];
  float ct[2];
  #pragma unroll
  for (int j = 0; j < 2; ++j) ct[j] = colterm[nwave + j * 16 + l16];
  float colm[2] = {-3.4e38f, -3.4e38f};

  #pragma unroll
  for (int i = 0; i < 4; ++i) {
    float rowm[8];
    #pragma unroll
    for (int r = 0; r < 8; ++r) {
      const float rt = rowterm[mwave + i * 16 + 8 * hf + r] + bcq;
      float mx = -3.4e38f;
      #pragma unroll
      for (int j = 0; j < 2; ++j) {
        const float s = acc[i][j][r] + rt + ct[j];   // C/D: M = r+8*half, N = lane%16
        mx = fmaxf(mx, s);
        colm[j] = fmaxf(colm[j], s);
      }
      rowm[r] = mx;
    }
    #pragma unroll
    for (int r = 0; r < 8; ++r) {
      float v = rowm[r];
      v = fmaxf(v, __shfl_xor(v, 1, 32));
      v = fmaxf(v, __shfl_xor(v, 2, 32));
      v = fmaxf(v, __shfl_xor(v, 4, 32));
      v = fmaxf(v, __shfl_xor(v, 8, 32));
      if (l16 == 0)
        atomicMax(&rowMaxU[mwave + i * 16 + 8 * hf + r], enc_ord(v));
    }
  }
  #pragma unroll
  for (int j = 0; j < 2; ++j) {
    float v = fmaxf(colm[j], __shfl_xor(colm[j], 16, 32));
    if (lane < 16)
      atomicMax(&colMaxU[nwave + j * 16 + l16], enc_ord(v));
  }
}

// ---- softmax over n encoded-max values (single block) ----
__global__ void softmax_kernel(const unsigned* __restrict__ mU,
                               float* __restrict__ w, int n) {
  __shared__ float red[256];
  const int t = threadIdx.x;
  float m = -3.4e38f;
  for (int i = t; i < n; i += 256) m = fmaxf(m, dec_ord(mU[i]));
  red[t] = m; __syncthreads();
  for (int s = 128; s > 0; s >>= 1) {
    if (t < s) red[t] = fmaxf(red[t], red[t + s]);
    __syncthreads();
  }
  m = red[0]; __syncthreads();
  float sum = 0.f;
  for (int i = t; i < n; i += 256) sum += expf(dec_ord(mU[i]) - m);
  red[t] = sum; __syncthreads();
  for (int s = 128; s > 0; s >>= 1) {
    if (t < s) red[t] += red[t + s];
    __syncthreads();
  }
  const float inv = 1.f / red[0];
  for (int i = t; i < n; i += 256) w[i] = expf(dec_ord(mU[i]) - m) * inv;
}

// ---- out[j] = scale * sum_i w[i] * M[i*stride + offc + j], one block per j ----
__global__ void wdot_kernel(const float* __restrict__ w,
                            const float* __restrict__ M,
                            int ni, int stride, int offc, float scale,
                            float* __restrict__ out) {
  __shared__ float red[256];
  const int t = threadIdx.x;
  const int j = blockIdx.x;
  float acc = 0.f;
  for (int i = t; i < ni; i += 256)
    acc += w[i] * M[(size_t)i * stride + offc + j];
  red[t] = acc; __syncthreads();
  for (int s = 128; s > 0; s >>= 1) {
    if (t < s) red[t] += red[t + s];
    __syncthreads();
  }
  if (t == 0) out[j] = scale * red[0];
}

// ---- batch-1 GEMV (y = act(x @ W + b)), coalesced over nout, x staged in LDS ----
__global__ void gemv_relu_kernel(const float* __restrict__ x,
                                 const float* __restrict__ Wm,
                                 const float* __restrict__ b,
                                 float* __restrict__ y,
                                 int nin, int nout, int dorelu) {
  __shared__ float xs[256];
  const int t = threadIdx.x;
  const int n = blockIdx.x * 256 + t;
  float acc = 0.f;
  for (int k0 = 0; k0 < nin; k0 += 256) {
    __syncthreads();
    const int k = k0 + t;
    xs[t] = (k < nin) ? x[k] : 0.f;
    __syncthreads();
    const int kend = (nin - k0 < 256) ? (nin - k0) : 256;
    if (n < nout) {
      for (int kk = 0; kk < kend; ++kk)
        acc = fmaf(xs[kk], Wm[(size_t)(k0 + kk) * nout + n], acc);
    }
  }
  if (n < nout) {
    acc += b[n];
    if (dorelu) acc = fmaxf(acc, 0.f);
    y[n] = acc;
  }
}

extern "C" void kernel_launch(void* const* d_in, const int* in_sizes, int n_in,
                              void* d_out, int out_size, void* d_ws, size_t ws_size,
                              hipStream_t stream) {
  (void)in_sizes; (void)n_in; (void)out_size; (void)ws_size;
  const float* title = (const float*)d_in[0];
  const float* data  = (const float*)d_in[1];
  const float* w_c   = (const float*)d_in[2];
  const float* b_c   = (const float*)d_in[3];
  const float* w_q   = (const float*)d_in[4];
  const float* b_q   = (const float*)d_in[5];
  const float* w_cq  = (const float*)d_in[6];
  const float* b_cq  = (const float*)d_in[7];
  const float* Wt[7]; const float* Bt[7];
  for (int i = 0; i < 7; ++i) {
    Wt[i] = (const float*)d_in[8 + 2 * i];
    Bt[i] = (const float*)d_in[9 + 2 * i];
  }

  size_t off = 0;
  auto carve = [&](size_t bytes) -> void* {
    void* p = (char*)d_ws + off;
    off = (off + bytes + 255) & ~(size_t)255;
    return p;
  };
  unsigned short* hwB  = (unsigned short*)carve((size_t)NUM_HEADERS * KPAD * 2);
  unsigned short* tbB  = (unsigned short*)carve((size_t)TITLE_LEN * KPAD * 2);
  float* rowterm  = (float*)carve((size_t)NUM_HEADERS * 4);
  float* colterm  = (float*)carve((size_t)TITLE_LEN * 4);
  unsigned* rowMaxU = (unsigned*)carve((size_t)NUM_HEADERS * 4);
  unsigned* colMaxU = (unsigned*)carve((size_t)TITLE_LEN * 4);
  float* col_w    = (float*)carve((size_t)NUM_HEADERS * 4);
  float* title_w  = (float*)carve((size_t)TITLE_LEN * 4);
  float* x0       = (float*)carve(512 * 4);
  float* yA       = (float*)carve(3072 * 4);
  float* yB       = (float*)carve(3072 * 4);

  prep_headers_kernel<<<NUM_HEADERS / 8, 256, 0, stream>>>(
      data, w_cq, w_c, b_c, hwB, rowterm, rowMaxU);
  prep_title_kernel<<<TITLE_LEN / 8, 256, 0, stream>>>(
      title, w_q, b_q, tbB, colterm, colMaxU);

  dim3 grid(TITLE_LEN / 128, NUM_HEADERS / 128);
  sim_gemm_max_kernel<<<grid, 256, 0, stream>>>(
      hwB, tbB, rowterm, colterm, b_cq, rowMaxU, colMaxU);

  softmax_kernel<<<1, 256, 0, stream>>>(rowMaxU, col_w, NUM_HEADERS);
  softmax_kernel<<<1, 256, 0, stream>>>(colMaxU, title_w, TITLE_LEN);

  // t2q = Q * col_w @ col_feat (data[:,100:500]); q2t = C * title_w @ title
  wdot_kernel<<<400, 256, 0, stream>>>(col_w, data, NUM_HEADERS, 500, 100,
                                       (float)NUM_HEADERS, x0);
  wdot_kernel<<<100, 256, 0, stream>>>(title_w, title, TITLE_LEN, 100, 0,
                                       (float)TITLE_LEN, x0 + 400);

  const int dims[8] = {500, 500, 1000, 3000, 1000, 500, 100, 8};
  const float* cur = x0;
  float* bufs[2] = {yA, yB};
  for (int i = 0; i < 7; ++i) {
    float* outp = (i == 6) ? (float*)d_out : bufs[i & 1];
    const int nin = dims[i], nout = dims[i + 1];
    gemv_relu_kernel<<<(nout + 255) / 256, 256, 0, stream>>>(
        cur, Wt[i], Bt[i], outp, nin, nout, (i == 0) ? 0 : 1);
    cur = outp;
  }
}
